// SimpleDIN_65738769433070
// MI455X (gfx1250) — matrix operand order repaired and checked by reading, not compiled, not run on hardware
//
#include <hip/hip_runtime.h>
#include <cmath>

typedef __attribute__((ext_vector_type(16))) _Float16 v16h;
typedef __attribute__((ext_vector_type(8)))  _Float16 v8h;
typedef __attribute__((ext_vector_type(8)))  float    v8f;

#define B_SZ   4096
#define S_SEQ  200
#define S_PAD  208     // 13 tiles of 16 rows
#define D_EMB  64
#define H_ATT  80
#define NT_N   5       // 80 / 16 n-tiles
#define KC_N   4       // reduced K = 128 -> 4 chunks of 32
#define MT_N   13      // 208 / 16 m-tiles

#define FRAG_HALVES (NT_N * KC_N * 512)       // 10240 f16 = 20 KB
#define WBD_FLOATS  (D_EMB * H_ATT)           // 5120 f32 = 20 KB

// ---------------------------------------------------------------------------
// Algebraic reduction: att_w1 = [Wa;Wb;Wc;Wd] over feature blocks
// [be, tgt, be*tgt, be-tgt].  Since tgt is constant per batch row:
//   h = be@(Wa+Wd) + (be*tgt)@Wc + (tgt@(Wb-Wd) + b1)
// -> WMMA K shrinks 256 -> 128 (2x FLOP cut); the bracketed bias is a tiny
// per-row 64x80 matvec done on VALU.
//
// Prep writes: (1) f16 B-fragments of W' = [Wa+Wd ; Wc] in wave32 WMMA
// B-layout (lane n<16: N=n, halves h -> K=kc*32+(h&7)+16*(h>>3); lane>=16: N=n-16,
// K=kc*32+8+(h&7)+16*(h>>3)), 32B contiguous per lane; (2) Wbd = Wb-Wd as f32.
// ---------------------------------------------------------------------------
__global__ __launch_bounds__(256) void din_prep_w1(const float* __restrict__ w1,
                                                   _Float16* __restrict__ frag,
                                                   float* __restrict__ wbd)
{
    int idx = blockIdx.x * 256 + threadIdx.x;
    if (idx < FRAG_HALVES) {
        int f    = idx >> 9;        // fragment id = nt*KC_N + kc
        int pos  = idx & 511;
        int lane = pos >> 4;
        int h    = pos & 15;
        int nt   = f / KC_N;
        int kc   = f % KC_N;
        int K = kc * 32 + ((lane >> 4) << 3) + (h & 7) + ((h >> 3) << 4);   // 0..127
        int N = nt * 16 + (lane & 15);
        float val = (K < 64)
                  ? w1[K * H_ATT + N] + w1[(192 + K) * H_ATT + N]   // Wa+Wd
                  : w1[(64 + K) * H_ATT + N];                       // Wc (K-64+128)
        frag[idx] = (_Float16)val;
    } else if (idx < FRAG_HALVES + WBD_FLOATS) {
        int j = idx - FRAG_HALVES;          // j = k*80 + n
        int k = j / H_ATT, n = j % H_ATT;
        wbd[j] = w1[(64 + k) * H_ATT + n] - w1[(192 + k) * H_ATT + n];
    }
}

__device__ __forceinline__ v16h cat8(v8h lo, v8h hi) {
    v16h r;
#pragma unroll
    for (int i = 0; i < 8; ++i) { r[i] = lo[i]; r[8 + i] = hi[i]; }
    return r;
}

// ---------------------------------------------------------------------------
// One workgroup (8 wave32) per batch row.
// ---------------------------------------------------------------------------
__global__ __launch_bounds__(256) void din_main(
    const int*   __restrict__ user_ids,
    const int*   __restrict__ item_ids,
    const int*   __restrict__ seqs,
    const float* __restrict__ user_table,
    const float* __restrict__ item_table,
    const float* __restrict__ att_b1,
    const float* __restrict__ att_w2,
    const float* __restrict__ pred_w1,
    const float* __restrict__ pred_b1,
    const float* __restrict__ pred_w2,
    const float* __restrict__ pred_b2,
    const float* __restrict__ pred_w3,
    const float* __restrict__ pred_b3,
    const _Float16* __restrict__ w1frag,
    const float* __restrict__ wbd,
    float* __restrict__ out)
{
    __shared__ __attribute__((aligned(16))) _Float16 s_be[S_PAD][D_EMB]; // 26.6 KB
    __shared__ __attribute__((aligned(16))) _Float16 s_tgt[D_EMB];
    __shared__ float s_tgt32[D_EMB];
    __shared__ float s_bias[H_ATT];          // tgt@(Wb-Wd) + b1
    __shared__ float s_scores[S_PAD];
    __shared__ float s_attn[S_SEQ];
    __shared__ float s_w2[H_ATT];
    __shared__ float s_red[8];
    __shared__ float s_pool[4][D_EMB];
    __shared__ float s_x[2 * D_EMB];
    __shared__ float s_h1[64];
    __shared__ float s_h2[32];

    const int b    = blockIdx.x;
    const int t    = threadIdx.x;
    const int lane = t & 31;
    const int wave = t >> 5;

    // ---- Phase A: gather embeddings (item table is L2-resident) ----
    const int* seq = seqs + (size_t)b * S_SEQ;
    for (int i = t; i < S_SEQ * D_EMB; i += 256) {
        int s = i >> 6, d = i & 63;
        s_be[s][d] = (_Float16)item_table[(size_t)seq[s] * D_EMB + d];
    }
    for (int i = t; i < (S_PAD - S_SEQ) * D_EMB; i += 256)
        s_be[S_SEQ + (i >> 6)][i & 63] = (_Float16)0.f;
    if (t < D_EMB) {
        float tv = item_table[(size_t)item_ids[b] * D_EMB + t];
        s_tgt[t]   = (_Float16)tv;
        s_tgt32[t] = tv;
        s_x[t]     = user_table[(size_t)user_ids[b] * D_EMB + t];
    }
    if (t < H_ATT) s_w2[t] = att_w2[t];
    __syncthreads();

    // ---- Per-row attention bias: bias[n] = b1[n] + sum_k tgt[k]*Wbd[k][n] ----
    if (t < H_ATT) {
        float a = att_b1[t];
        for (int k = 0; k < D_EMB; ++k) a += s_tgt32[k] * wbd[k * H_ATT + t];
        s_bias[t] = a;
    }
    __syncthreads();

    // ---- Phase B: fused [be, be*tgt](128) @ W' -> +bias -> ReLU -> @ att_w2 ----
    // A layout (16-bit 16x32): lanes 0-15 M=lane,  K {c..c+7, c+16..c+23}
    //                          lanes 16-31 M=lane-16, K shifted by +8.
    const int  lane_lo = lane & 15;
    const bool hiHalf  = lane >= 16;
    for (int mt = wave; mt < MT_N; mt += 8) {
        v8f acc[NT_N] = {};
        const int srow = mt * 16 + lane_lo;
#pragma unroll
        for (int kc = 0; kc < KC_N; ++kc) {
            const int c0 = ((kc & 1) << 5) + (hiHalf ? 8 : 0);
            v8h be_lo = *(const v8h*)&s_be[srow][c0];
            v8h be_hi = *(const v8h*)&s_be[srow][c0 + 16];
            v16h a;
            if (kc < 2) {
                a = cat8(be_lo, be_hi);                       // block 0: be
            } else {
                v8h t_lo = *(const v8h*)&s_tgt[c0];
                v8h t_hi = *(const v8h*)&s_tgt[c0 + 16];
                a = cat8(be_lo * t_lo, be_hi * t_hi);         // block 1: be*tgt
            }
#pragma unroll
            for (int nt = 0; nt < NT_N; ++nt) {
                v16h bf = *(const v16h*)&w1frag[(size_t)((nt * KC_N + kc) * 32 + lane) * 16];
                acc[nt] = __builtin_amdgcn_wmma_f32_16x16x32_f16(
                    false, a, false, bf, (short)0, acc[nt], false, false);
            }
        }
        // Epilogue: score[M] = sum_N relu(h[M,N] + bias[N]) * w2[N]
        // D layout: lanes 0-15 row M=mt*16+r, lanes 16-31 row M=mt*16+8+r
#pragma unroll
        for (int r = 0; r < 8; ++r) {
            float sc = 0.f;
#pragma unroll
            for (int nt = 0; nt < NT_N; ++nt) {
                int N = nt * 16 + lane_lo;
                sc += fmaxf(acc[nt][r] + s_bias[N], 0.f) * s_w2[N];
            }
            sc += __shfl_xor(sc, 1);
            sc += __shfl_xor(sc, 2);
            sc += __shfl_xor(sc, 4);
            sc += __shfl_xor(sc, 8);
            if (lane_lo == 0)
                s_scores[mt * 16 + (hiHalf ? 8 : 0) + r] = sc;
        }
    }
    __syncthreads();

    // ---- Phase C: softmax over 200 (padded rows masked; att_b2 shift-invariant) ----
    float v = (t < S_SEQ) ? s_scores[t] : -1e30f;
    float m = v;
#pragma unroll
    for (int off = 16; off > 0; off >>= 1) m = fmaxf(m, __shfl_xor(m, off));
    if (lane == 0) s_red[wave] = m;
    __syncthreads();
    if (t == 0) {
        float mm = s_red[0];
        for (int i = 1; i < 8; ++i) mm = fmaxf(mm, s_red[i]);
        s_red[0] = mm;
    }
    __syncthreads();
    m = s_red[0];
    float e = (t < S_SEQ) ? __expf(v - m) : 0.f;
    __syncthreads();                 // everyone has read s_red[0]
    float sum = e;
#pragma unroll
    for (int off = 16; off > 0; off >>= 1) sum += __shfl_xor(sum, off);
    if (lane == 0) s_red[wave] = sum;
    __syncthreads();
    if (t == 0) {
        float tot = 0.f;
        for (int i = 0; i < 8; ++i) tot += s_red[i];
        s_red[0] = tot;
    }
    __syncthreads();
    float inv = 1.f / s_red[0];
    if (t < S_SEQ) s_attn[t] = e * inv;
    __syncthreads();

    // ---- Phase D: pooled[d] = sum_s attn[s] * be[s][d] ----
    {
        int d = t & 63, part = t >> 6;
        float p = 0.f;
        int s0 = part * 50;
        for (int s = s0; s < s0 + 50; ++s) p += s_attn[s] * (float)s_be[s][d];
        s_pool[part][d] = p;
    }
    __syncthreads();
    if (t < D_EMB)
        s_x[D_EMB + t] = s_pool[0][t] + s_pool[1][t] + s_pool[2][t] + s_pool[3][t];
    __syncthreads();

    // ---- Phase E: prediction MLP 128 -> 64 -> 32 -> 1 (weights L2-hot) ----
    if (t < 64) {
        float a = pred_b1[t];
        for (int k = 0; k < 128; ++k) a += s_x[k] * pred_w1[k * 64 + t];
        s_h1[t] = fmaxf(a, 0.f);
    }
    __syncthreads();
    if (t < 32) {
        float a = pred_b2[t];
        for (int k = 0; k < 64; ++k) a += s_h1[k] * pred_w2[k * 32 + t];
        s_h2[t] = fmaxf(a, 0.f);
    }
    __syncthreads();
    if (t == 0) {
        float a = pred_b3[0];
        for (int k = 0; k < 32; ++k) a += s_h2[k] * pred_w3[k];
        out[b] = 1.f / (1.f + __expf(-a));
    }
}

extern "C" void kernel_launch(void* const* d_in, const int* in_sizes, int n_in,
                              void* d_out, int out_size, void* d_ws, size_t ws_size,
                              hipStream_t stream)
{
    (void)in_sizes; (void)n_in; (void)out_size; (void)ws_size;
    const int*   user_ids   = (const int*)d_in[0];
    const int*   item_ids   = (const int*)d_in[1];
    const int*   seqs       = (const int*)d_in[2];
    const float* user_table = (const float*)d_in[3];
    const float* item_table = (const float*)d_in[4];
    const float* att_w1     = (const float*)d_in[5];
    const float* att_b1     = (const float*)d_in[6];
    const float* att_w2     = (const float*)d_in[7];
    // d_in[8] = att_b2: constant shift of scores, softmax-invariant -> unused
    const float* pred_w1    = (const float*)d_in[9];
    const float* pred_b1    = (const float*)d_in[10];
    const float* pred_w2    = (const float*)d_in[11];
    const float* pred_b2    = (const float*)d_in[12];
    const float* pred_w3    = (const float*)d_in[13];
    const float* pred_b3    = (const float*)d_in[14];

    _Float16* w1frag = (_Float16*)d_ws;                      // 20 KB f16 fragments
    float*    wbd    = (float*)((char*)d_ws + FRAG_HALVES * sizeof(_Float16)); // 20 KB f32
    float*    out    = (float*)d_out;

    int prep_elems = FRAG_HALVES + WBD_FLOATS;
    din_prep_w1<<<(prep_elems + 255) / 256, 256, 0, stream>>>(att_w1, w1frag, wbd);
    din_main<<<B_SZ, 256, 0, stream>>>(user_ids, item_ids, seqs, user_table, item_table,
                                       att_b1, att_w2, pred_w1, pred_b1, pred_w2,
                                       pred_b2, pred_w3, pred_b3, w1frag, wbd, out);
}
